// LstmNetTest_62036507623966
// MI455X (gfx1250) — matrix operand order repaired and checked by reading, not compiled, not run on hardware
//
#include <hip/hip_runtime.h>
#include <hip/hip_bf16.h>
#include <math.h>

// ---------------------------------------------------------------------------
// B=8, S=32, H=16.  Big GEMM: [256 x 346112] @ [346112 x 64] -> xg1 [256 x 64]
// Roofline: x streamed once = 354 MB -> ~15us @ 23.3 TB/s; Wih1 (88.6 MB)
// resides in the 192 MB L2.  Compute: v_wmma_f32_16x16x32_bf16 (f32 accum),
// needs ~0.75 PFLOP/s effective -- ample headroom on the bf16 WMMA path.
// Split-K: wave = 32 rows x 64 cols over 1664-deep K slice; 8 waves/block
// cover a 13312-deep K group; cross-wave LDS reduction, then f32 atomics.
// ---------------------------------------------------------------------------

typedef __attribute__((ext_vector_type(16))) __bf16 v16bf;
typedef __attribute__((ext_vector_type(8)))  float  v8f;

#define K1       346112   // 128*52*52
#define KWAVE    1664     // K slice per wave (52 iters of 32)
#define KGROUP   13312    // 8 * KWAVE, K per block
#define NKG      26       // K1 / KGROUP
#define NMG      8        // 256 rows / 32 rows per block
#define NM       256      // B*S rows
#define NG       64       // 4*H gates

__device__ __forceinline__ v16bf cvt16(const float* s) {
  v16bf r;
#pragma unroll
  for (int i = 0; i < 16; ++i) r[i] = (__bf16)s[i];
  return r;
}

// ---------------------------------------------------------------------------
// Kernel A: init xg1 with (bih1+bhh1); compute xg2 = l@Wih2^T + bih2 + bhhh2.
// ---------------------------------------------------------------------------
__global__ __launch_bounds__(256) void lstm_init_kernel(
    const float* __restrict__ l, const float* __restrict__ Wih2,
    const float* __restrict__ bih1, const float* __restrict__ bhh1,
    const float* __restrict__ bih2, const float* __restrict__ bhh2,
    float* __restrict__ xg1, float* __restrict__ xg2) {
  int i = blockIdx.x * blockDim.x + threadIdx.x;  // 0 .. 16383
  int g = i & 63;
  int r = i >> 6;
  xg1[i] = bih1[g] + bhh1[g];
  float acc = bih2[g] + bhh2[g];
#pragma unroll
  for (int k = 0; k < 5; ++k) acc += l[r * 5 + k] * Wih2[g * 5 + k];
  xg2[i] = acc;
}

// ---------------------------------------------------------------------------
// Kernel B: split-K bf16 WMMA GEMM with block-level LDS reduction.
// A frag (16x32, contiguous k, same as B):            lane half h -> K {16h..16h+15}, M=lane&15.
// B frag (32x16):                      lane half h -> K {16h..16h+15}, N=lane&15.
// C/D frag: lane half h, N = lane&15;  VGPR j -> M = 8h + j.
// ---------------------------------------------------------------------------
__global__ __launch_bounds__(256) void big_gemm_kernel(
    const float* __restrict__ X, const float* __restrict__ W,
    float* __restrict__ xg1) {
  __shared__ float red[8 * 2048];       // 8 waves x (8 tiles x 256) = 64 KB

  const int mgroup = blockIdx.x & 7;          // 0..7 : rows mgroup*32 ..+31
  const int kgroup = blockIdx.x >> 3;         // 0..25
  const int w      = threadIdx.x >> 5;        // wave in block, 0..7
  const int lane   = threadIdx.x & 31;
  const int half   = lane >> 4;
  const int r      = lane & 15;

  const long kbase = (long)kgroup * KGROUP + (long)w * KWAVE;

  const float* A0 = X + (long)(mgroup * 32 +      r) * K1;  // mtile 0
  const float* A1 = X + (long)(mgroup * 32 + 16 + r) * K1;  // mtile 1
  const float* B0 = W + (long)( 0 + r) * K1;
  const float* B1 = W + (long)(16 + r) * K1;
  const float* B2 = W + (long)(32 + r) * K1;
  const float* B3 = W + (long)(48 + r) * K1;

  v8f acc00 = {}, acc01 = {}, acc02 = {}, acc03 = {};
  v8f acc10 = {}, acc11 = {}, acc12 = {}, acc13 = {};

  for (long k = kbase; k < kbase + KWAVE; k += 32) {
    // stream-ahead prefetch of x (HBM-resident)
    __builtin_prefetch(A0 + k + 2048, 0, 0);
    __builtin_prefetch(A1 + k + 2048, 0, 0);

    // ---- stage ALL loads first so they issue as one deep clause ----
    const long ka = k + half * 16;      // A element run
    const long kb = k + half * 16;      // B element run
    float4 sa[8];
    sa[0] = *(const float4*)(A0 + ka);
    sa[1] = *(const float4*)(A0 + ka + 4);
    sa[2] = *(const float4*)(A0 + ka + 8);
    sa[3] = *(const float4*)(A0 + ka + 12);
    sa[4] = *(const float4*)(A1 + ka);
    sa[5] = *(const float4*)(A1 + ka + 4);
    sa[6] = *(const float4*)(A1 + ka + 8);
    sa[7] = *(const float4*)(A1 + ka + 12);
    float4 sb[16];
#pragma unroll
    for (int q = 0; q < 4; ++q) sb[0 + q]  = *(const float4*)(B0 + kb + q * 4);
#pragma unroll
    for (int q = 0; q < 4; ++q) sb[4 + q]  = *(const float4*)(B1 + kb + q * 4);
#pragma unroll
    for (int q = 0; q < 4; ++q) sb[8 + q]  = *(const float4*)(B2 + kb + q * 4);
#pragma unroll
    for (int q = 0; q < 4; ++q) sb[12 + q] = *(const float4*)(B3 + kb + q * 4);

    // ---- convert to bf16 fragments ----
    v16bf a0  = cvt16((const float*)&sa[0]);
    v16bf a1  = cvt16((const float*)&sa[4]);
    v16bf bf0 = cvt16((const float*)&sb[0]);
    v16bf bf1 = cvt16((const float*)&sb[4]);
    v16bf bf2 = cvt16((const float*)&sb[8]);
    v16bf bf3 = cvt16((const float*)&sb[12]);

    acc00 = __builtin_amdgcn_wmma_f32_16x16x32_bf16(false, a0, false, bf0,
                                                    (short)0, acc00, false, false);
    acc01 = __builtin_amdgcn_wmma_f32_16x16x32_bf16(false, a0, false, bf1,
                                                    (short)0, acc01, false, false);
    acc02 = __builtin_amdgcn_wmma_f32_16x16x32_bf16(false, a0, false, bf2,
                                                    (short)0, acc02, false, false);
    acc03 = __builtin_amdgcn_wmma_f32_16x16x32_bf16(false, a0, false, bf3,
                                                    (short)0, acc03, false, false);
    acc10 = __builtin_amdgcn_wmma_f32_16x16x32_bf16(false, a1, false, bf0,
                                                    (short)0, acc10, false, false);
    acc11 = __builtin_amdgcn_wmma_f32_16x16x32_bf16(false, a1, false, bf1,
                                                    (short)0, acc11, false, false);
    acc12 = __builtin_amdgcn_wmma_f32_16x16x32_bf16(false, a1, false, bf2,
                                                    (short)0, acc12, false, false);
    acc13 = __builtin_amdgcn_wmma_f32_16x16x32_bf16(false, a1, false, bf3,
                                                    (short)0, acc13, false, false);
  }

  // ---- spill accumulators to LDS: index = w*2048 + tile*256 + j*32 + lane ---
  float* my = &red[w * 2048 + lane];
#pragma unroll
  for (int j = 0; j < 8; ++j) {
    my[(0 * 256) + j * 32] = acc00[j];
    my[(1 * 256) + j * 32] = acc01[j];
    my[(2 * 256) + j * 32] = acc02[j];
    my[(3 * 256) + j * 32] = acc03[j];
    my[(4 * 256) + j * 32] = acc10[j];
    my[(5 * 256) + j * 32] = acc11[j];
    my[(6 * 256) + j * 32] = acc12[j];
    my[(7 * 256) + j * 32] = acc13[j];
  }
  __syncthreads();

  // ---- cross-wave reduce (8-way) then one pass of f32 atomics ----
#pragma unroll
  for (int p = threadIdx.x; p < 2048; p += 256) {
    float s = 0.0f;
#pragma unroll
    for (int ww = 0; ww < 8; ++ww) s += red[ww * 2048 + p];
    int tile  = p >> 8;          // mt*4 + nt
    int j     = (p >> 5) & 7;
    int l5    = p & 31;
    int h5    = l5 >> 4;
    int rr    = l5 & 15;
    int mt    = tile >> 2;
    int nt    = tile & 3;
    int row   = mgroup * 32 + mt * 16 + h5 * 8 + j;
    int col   = nt * 16 + rr;
    unsafeAtomicAdd(xg1 + row * NG + col, s);
  }
}

// ---------------------------------------------------------------------------
// Kernel C: both LSTM recurrences (S=32 sequential) + fused output FC.
// ---------------------------------------------------------------------------
__global__ __launch_bounds__(512) void recur_kernel(
    const float* __restrict__ xg1, const float* __restrict__ xg2,
    const float* __restrict__ Whh1, const float* __restrict__ Whh2,
    const float* __restrict__ fcW, const float* __restrict__ fcb,
    float* __restrict__ out) {
  __shared__ float sWhh[2][64 * 16];
  __shared__ float sfcW[64];
  __shared__ float h[2][8][16];
  __shared__ float c[2][8][16];
  __shared__ float gbuf[2][8][64];

  const int tid = threadIdx.x;
  for (int i = tid; i < 1024; i += 512) {
    sWhh[0][i] = Whh1[i];
    sWhh[1][i] = Whh2[i];
  }
  if (tid < 64) sfcW[tid] = fcW[tid];
  if (tid < 256) {
    ((float*)h)[tid] = 0.0f;
    ((float*)c)[tid] = 0.0f;
  }
  __syncthreads();

  for (int t = 0; t < 32; ++t) {
#pragma unroll
    for (int rep = 0; rep < 2; ++rep) {
      int idx  = tid + rep * 512;
      int lstm = idx >> 9;
      int b    = (idx >> 6) & 7;
      int g    = idx & 63;
      const float* xg = lstm ? xg2 : xg1;
      float v = xg[(b * 32 + t) * 64 + g];
      const float* wrow = &sWhh[lstm][g * 16];
      const float* hb   = &h[lstm][b][0];
#pragma unroll
      for (int j = 0; j < 16; ++j) v += hb[j] * wrow[j];
      gbuf[lstm][b][g] = v;
    }
    __syncthreads();

    if (tid < 256) {
      int lstm = tid >> 7;
      int b    = (tid >> 4) & 7;
      int j    = tid & 15;
      const float* gb = &gbuf[lstm][b][0];
      float ig = 1.0f / (1.0f + expf(-gb[j]));
      float fg = 1.0f / (1.0f + expf(-gb[16 + j]));
      float gg = tanhf(gb[32 + j]);
      float og = 1.0f / (1.0f + expf(-gb[48 + j]));
      float cn = fg * c[lstm][b][j] + ig * gg;
      c[lstm][b][j] = cn;
      h[lstm][b][j] = og * tanhf(cn);
    }
    __syncthreads();

    if (tid < 32) {
      int b = tid >> 2;
      int o = tid & 3;
      float acc = fcb[o];
#pragma unroll
      for (int j = 0; j < 16; ++j)
        acc += (h[0][b][j] + h[1][b][j]) * sfcW[o * 16 + j];
      out[(b * 32 + t) * 4 + o] = acc;
    }
    __syncthreads();
  }
}

// ---------------------------------------------------------------------------
extern "C" void kernel_launch(void* const* d_in, const int* in_sizes, int n_in,
                              void* d_out, int out_size, void* d_ws, size_t ws_size,
                              hipStream_t stream) {
  const float* x    = (const float*)d_in[0];
  const float* l    = (const float*)d_in[1];
  const float* Wih1 = (const float*)d_in[2];
  const float* Whh1 = (const float*)d_in[3];
  const float* bih1 = (const float*)d_in[4];
  const float* bhh1 = (const float*)d_in[5];
  const float* Wih2 = (const float*)d_in[6];
  const float* Whh2 = (const float*)d_in[7];
  const float* bih2 = (const float*)d_in[8];
  const float* bhh2 = (const float*)d_in[9];
  const float* fcW  = (const float*)d_in[10];
  const float* fcb  = (const float*)d_in[11];

  float* xg1 = (float*)d_ws;            // [256 x 64]
  float* xg2 = xg1 + NM * NG;           // [256 x 64]
  float* out = (float*)d_out;           // [8 x 32 x 4]

  lstm_init_kernel<<<64, 256, 0, stream>>>(l, Wih2, bih1, bhh1, bih2, bhh2,
                                           xg1, xg2);
  // 8 mgroups x 26 kgroups = 208 blocks of 8 waves (1664 waves total)
  big_gemm_kernel<<<NMG * NKG, 256, 0, stream>>>(x, Wih1, xg1);
  recur_kernel<<<1, 512, 0, stream>>>(xg1, xg2, Whh1, Whh2, fcW, fcb, out);
}
